// GATEncoderGraph_5738076307608
// MI455X (gfx1250) — compile-verified, matrix operand order checked by reading, not run on hardware
//
#include <hip/hip_runtime.h>
#include <hip/hip_bf16.h>

// GAT encoder: B=8, N=1024, F_IN=F_HID=F_EMB=128, H=8, L_DIM=2
#define BB 8
#define NN 1024
#define FF 128
#define HH 8
#define PRED_IN 384
#define LDIM 2

typedef __attribute__((ext_vector_type(16))) __bf16 v16bf;
typedef __attribute__((ext_vector_type(8)))  __bf16 v8bf;
typedef __attribute__((ext_vector_type(8)))  float  v8f;
typedef __attribute__((ext_vector_type(4)))  float  v4f;

__device__ __forceinline__ __bf16 f2bf(float f) { return (__bf16)f; }

// Async global->LDS 16B copy (ASYNCcnt-tracked, ISA 15.18.3 op 98).
// LDS address = low 32 bits of the flat pointer (ISA 10.2 LDS aperture).
__device__ __forceinline__ void async_copy_b128(const void* g, void* l) {
  unsigned loff = (unsigned)(unsigned long long)l;
  asm volatile("global_load_async_to_lds_b128 %0, %1, off"
               :: "v"(loff), "v"(g) : "memory");
}
__device__ __forceinline__ void wait_async0() {
  asm volatile("s_wait_asynccnt 0" ::: "memory");
}

// ---------------- cast f32 -> bf16 (row-major copy) ----------------
__global__ void cast_f32_bf16_kernel(const float* __restrict__ in,
                                     __bf16* __restrict__ out, int n) {
  int i = blockIdx.x * blockDim.x + threadIdx.x;
  if (i < n) out[i] = f2bf(in[i]);
}

// ---------------- transpose-cast weights: w[h,f,o] -> wbfT[h,o,f] -----------
__global__ void cast_w_transpose_kernel(const float* __restrict__ w,
                                        __bf16* __restrict__ wbfT) {
  int tid = blockIdx.x * blockDim.x + threadIdx.x;   // H*FF*FF, f fastest
  if (tid >= HH * FF * FF) return;
  int f = tid & 127;
  int o1 = (tid >> 7) & 127;
  int h = tid >> 14;
  wbfT[(h * FF + o1) * FF + f] = f2bf(w[(h * FF + f) * FF + o1]);
}

// ---------------- hp[b,h] = x[b] @ w[h] (bf16 WMMA, transposed output) ------
__global__ void gemm_xw_kernel(const __bf16* __restrict__ xbf,   // [B,N,128]
                               const __bf16* __restrict__ wbfT,  // [H,128,128] (o,f)
                               __bf16* __restrict__ hpbfT) {     // [B*H,128,N]
  int gtid = blockIdx.x * blockDim.x + threadIdx.x;
  int wid  = gtid >> 5;
  int lane = gtid & 31;
  int nt = wid & 7;
  int mt = (wid >> 3) & 63;
  int h  = (wid >> 9) & 7;
  int b  = (wid >> 12) & 7;

  int half = lane >> 4;
  int r15  = lane & 15;
  int arow = mt * 16 + r15;
  int col  = nt * 16 + r15;

  const __bf16* xrow = xbf + ((size_t)(b * NN + arow)) * FF;
  const __bf16* wcol = wbfT + ((size_t)(h * FF + col)) * FF;

  v8f c = {};
#pragma unroll
  for (int kc = 0; kc < 4; ++kc) {
    int k0 = kc * 32;
    const __bf16* ap = xrow + k0 + half * 8;
    v8bf alo = *(const v8bf*)(ap);
    v8bf ahi = *(const v8bf*)(ap + 16);
    v16bf a = __builtin_shufflevector(alo, ahi, 0, 1, 2, 3, 4, 5, 6, 7,
                                      8, 9, 10, 11, 12, 13, 14, 15);
    const __bf16* bp = wcol + k0 + half * 8;
    v8bf blo = *(const v8bf*)(bp);
    v8bf bhi = *(const v8bf*)(bp + 16);
    v16bf bfr = __builtin_shufflevector(blo, bhi, 0, 1, 2, 3, 4, 5, 6, 7,
                                        8, 9, 10, 11, 12, 13, 14, 15);
    c = __builtin_amdgcn_wmma_f32_16x16x32_bf16(false, a, false, bfr,
                                                (short)0, c, false, false);
  }

  v8bf cb;
#pragma unroll
  for (int r = 0; r < 8; ++r) cb[r] = f2bf(c[r]);
  size_t bh = (size_t)(b * HH + h);
  __bf16* dst = hpbfT + (bh * FF + col) * NN + mt * 16 + half * 8;
  *(v8bf*)dst = cb;
}

// ---------------- s,d = tanh(hp) . a_src/a_dst (reads transposed hp) --------
__global__ void sd_kernel(const __bf16* __restrict__ hpbfT,   // [BH,128,N]
                          const float* __restrict__ asrc,     // [H,128]
                          const float* __restrict__ adst,     // [H,128]
                          float* __restrict__ s, float* __restrict__ d) {
  int tid = blockIdx.x * blockDim.x + threadIdx.x;   // B*H*N
  if (tid >= BB * HH * NN) return;
  int n  = tid & 1023;
  int bh = tid >> 10;
  int h  = bh & 7;
  const __bf16* base = hpbfT + (size_t)bh * FF * NN + n;
  const float* as = asrc + h * FF;
  const float* ad = adst + h * FF;
  float sv = 0.f, dv = 0.f;
#pragma unroll 4
  for (int o = 0; o < FF; ++o) {
    float t = tanhf((float)base[(size_t)o * NN]);
    sv += t * as[o];
    dv += t * ad[o];
  }
  s[tid] = sv; d[tid] = dv;
}

// ---------------- per-row softmax stats (max, sum-exp) over masked scores ----
__global__ void rowstats_kernel(const float* __restrict__ s,
                                const float* __restrict__ d,
                                const float* __restrict__ adj,   // [B,N,N]
                                float* __restrict__ rmax,
                                float* __restrict__ rsum) {
  int gtid = blockIdx.x * blockDim.x + threadIdx.x;
  int wid  = gtid >> 5;
  int lane = gtid & 31;
  int n = wid & 1023;
  int h = (wid >> 10) & 7;
  int b = (wid >> 13) & 7;
  int idx = (b * HH + h) * NN + n;
  float sv = s[idx];
  const float* arow = adj + ((size_t)(b * NN + n)) * NN;
  const float* dh   = d + (size_t)(b * HH + h) * NN;

  float mx = -3.0e38f;
  for (int m = lane; m < NN; m += 32) {
    if (arow[m] > 0.f) {
      float sc = sv + dh[m];
      sc = (sc >= 0.f) ? sc : 0.2f * sc;
      mx = fmaxf(mx, sc);
    }
  }
#pragma unroll
  for (int off = 16; off > 0; off >>= 1) mx = fmaxf(mx, __shfl_xor(mx, off, 32));

  float sm = 0.f;
  for (int m = lane; m < NN; m += 32) {
    if (arow[m] > 0.f) {
      float sc = sv + dh[m];
      sc = (sc >= 0.f) ? sc : 0.2f * sc;
      sm += __expf(sc - mx);
    }
  }
#pragma unroll
  for (int off = 16; off > 0; off >>= 1) sm += __shfl_xor(sm, off, 32);

  if (lane == 0) { rmax[idx] = mx; rsum[idx] = sm; }
}

// ---------------- fused softmax + (attn @ hp) via bf16 WMMA -----------------
// Block = 4 waves sharing one (b,h). The 32x128 bf16 hp slab for each
// neighbor chunk (8 KB) is staged global->LDS with async b128 copies,
// double-buffered (copy of chunk i+1 overlaps the 8 WMMAs of chunk i).
// B-fragments then come from LDS as 2 x 16B ds reads per tile.
__global__ void aggregate_kernel(const float* __restrict__ s,
                                 const float* __restrict__ d,
                                 const float* __restrict__ rmax,
                                 const float* __restrict__ rsum,
                                 const float* __restrict__ adj,     // [B,N,N]
                                 const __bf16* __restrict__ hpbfT,  // [BH,128,N]
                                 float* __restrict__ outbhT) {      // [BH,128,N]
  __shared__ __align__(16) __bf16 slab[2][FF * 32];   // 2 x 8 KB

  int tid  = threadIdx.x;                  // 0..127
  int lane = tid & 31;
  int gwid = blockIdx.x * 4 + (tid >> 5);  // 0..4095; 4 waves share (b,h)
  int nt = gwid & 63;
  int h  = (gwid >> 6) & 7;
  int b  = (gwid >> 9) & 7;

  int half = lane >> 4;
  int r15  = lane & 15;
  int arow = nt * 16 + r15;

  size_t bhbase = (size_t)(b * HH + h) * NN;
  float sv  = s[bhbase + arow];
  float mx  = rmax[bhbase + arow];
  float inv = 1.0f / rsum[bhbase + arow];
  const float* arowp = adj + ((size_t)(b * NN + arow)) * NN;
  const float* dh    = d + bhbase;
  const __bf16* hpb  = hpbfT + bhbase * FF;   // [128 cols][NN rows]

  v8f acc[8];
#pragma unroll
  for (int t = 0; t < 8; ++t) acc[t] = (v8f){};

  // stage chunk mc's 128x32 slab: 512 pieces of 16B; 4 per thread
  auto stage = [&](int mc, int bufsel) {
#pragma unroll
    for (int k = 0; k < 4; ++k) {
      int piece = k * 128 + tid;        // 0..511
      int c  = piece >> 2;              // column 0..127
      int r0 = (piece & 3) * 8;         // row group within chunk
      async_copy_b128(hpb + (size_t)c * NN + mc + r0,
                      &slab[bufsel][piece * 8]);
    }
  };

  stage(0, 0);

  for (int mc = 0; mc < NN; mc += 32) {
    int pb = (mc >> 5) & 1;
    wait_async0();        // my async copies for chunk mc are done
    __syncthreads();      // everyone's copies done; prev buffer free
    if (mc + 32 < NN) stage(mc + 32, pb ^ 1);

    __builtin_prefetch(arowp + mc + 32, 0, 0);
    // attention A-fragment from float4 adj/d loads (ISA A-map groups)
    const float* ap0 = arowp + mc + half * 8;
    const float* ap1 = arowp + mc + 16 + half * 8;
    const float* dp0 = dh + mc + half * 8;
    const float* dp1 = dh + mc + 16 + half * 8;
    v4f A0 = *(const v4f*)(ap0);
    v4f A1 = *(const v4f*)(ap0 + 4);
    v4f A2 = *(const v4f*)(ap1);
    v4f A3 = *(const v4f*)(ap1 + 4);
    v4f D0 = *(const v4f*)(dp0);
    v4f D1 = *(const v4f*)(dp0 + 4);
    v4f D2 = *(const v4f*)(dp1);
    v4f D3 = *(const v4f*)(dp1 + 4);

    float p[16];
#pragma unroll
    for (int j = 0; j < 4; ++j) {
      float sc;
      sc = sv + D0[j]; sc = (sc >= 0.f) ? sc : 0.2f * sc;
      p[j]      = (A0[j] > 0.f) ? __expf(sc - mx) * inv : 0.f;
      sc = sv + D1[j]; sc = (sc >= 0.f) ? sc : 0.2f * sc;
      p[j + 4]  = (A1[j] > 0.f) ? __expf(sc - mx) * inv : 0.f;
      sc = sv + D2[j]; sc = (sc >= 0.f) ? sc : 0.2f * sc;
      p[j + 8]  = (A2[j] > 0.f) ? __expf(sc - mx) * inv : 0.f;
      sc = sv + D3[j]; sc = (sc >= 0.f) ? sc : 0.2f * sc;
      p[j + 12] = (A3[j] > 0.f) ? __expf(sc - mx) * inv : 0.f;
    }
    v16bf a;
#pragma unroll
    for (int j = 0; j < 16; ++j) a[j] = f2bf(p[j]);

#pragma unroll
    for (int ct = 0; ct < 8; ++ct) {
      int col = ct * 16 + r15;
      const __bf16* sp = &slab[pb][col * 32];
      v8bf blo = *(const v8bf*)(sp + half * 8);
      v8bf bhi = *(const v8bf*)(sp + 16 + half * 8);
      v16bf bfr = __builtin_shufflevector(blo, bhi, 0, 1, 2, 3, 4, 5, 6, 7,
                                          8, 9, 10, 11, 12, 13, 14, 15);
      acc[ct] = __builtin_amdgcn_wmma_f32_16x16x32_bf16(false, a, false, bfr,
                                                        (short)0, acc[ct],
                                                        false, false);
    }
  }

  // transposed store: per col-tile, lane's 8 rows are contiguous -> 2 x b128
#pragma unroll
  for (int ct = 0; ct < 8; ++ct) {
    int col = ct * 16 + r15;
    float* dst = outbhT + ((size_t)(b * HH + h) * FF + col) * NN
               + nt * 16 + half * 8;
    v4f lo = __builtin_shufflevector(acc[ct], acc[ct], 0, 1, 2, 3);
    v4f hi = __builtin_shufflevector(acc[ct], acc[ct], 4, 5, 6, 7);
    *(v4f*)(dst)     = lo;
    *(v4f*)(dst + 4) = hi;
  }
}

// ---------------- mean over heads + bias (+ReLU), emit f32 & bf16 -----------
__global__ void headmean_kernel(const float* __restrict__ outbhT, // [BH,128,N]
                                const float* __restrict__ bias,   // [128]
                                float* __restrict__ xout,         // [B,N,128]
                                __bf16* __restrict__ xbf,         // [B,N,128]
                                int do_relu) {
  int tid = blockIdx.x * blockDim.x + threadIdx.x;   // B*128*N, n fastest
  if (tid >= BB * NN * FF) return;
  int n = tid & 1023;
  int o = (tid >> 10) & 127;
  int b = tid >> 17;
  float v = 0.f;
#pragma unroll
  for (int h = 0; h < HH; ++h)
    v += outbhT[((size_t)(b * HH + h) * FF + o) * NN + n];
  v = v * (1.0f / HH) + bias[o];
  if (do_relu) v = fmaxf(v, 0.f);
  size_t didx = ((size_t)(b * NN + n)) * FF + o;   // row-major for next GEMM A
  xout[didx] = v;
  xbf[didx]  = f2bf(v);
}

// ---------------- max-pool over nodes of concat(x1,x2,x3) -------------------
__global__ void pool_kernel(const float* __restrict__ x1,
                            const float* __restrict__ x2,
                            const float* __restrict__ x3,
                            float* __restrict__ pooled) {   // [B, 384]
  int tid = blockIdx.x * blockDim.x + threadIdx.x;
  if (tid >= BB * PRED_IN) return;
  int b = tid / PRED_IN;
  int c = tid % PRED_IN;
  const float* src; int cc;
  if (c < 128)      { src = x1; cc = c; }
  else if (c < 256) { src = x2; cc = c - 128; }
  else              { src = x3; cc = c - 256; }
  float mx = -3.0e38f;
  for (int n = 0; n < NN; ++n)
    mx = fmaxf(mx, src[((size_t)(b * NN + n)) * FF + cc]);
  pooled[tid] = mx;
}

// ---------------- final tiny head: pooled @ pw + pb -------------------------
__global__ void predict_kernel(const float* __restrict__ pooled,
                               const float* __restrict__ pw,   // [384,2]
                               const float* __restrict__ pb,   // [2]
                               float* __restrict__ out) {      // [B,2]
  int tid = blockIdx.x * blockDim.x + threadIdx.x;
  if (tid >= BB * LDIM) return;
  int b = tid / LDIM;
  int j = tid % LDIM;
  float acc = pb[j];
  for (int c = 0; c < PRED_IN; ++c)
    acc += pooled[b * PRED_IN + c] * pw[c * LDIM + j];
  out[tid] = acc;
}

extern "C" void kernel_launch(void* const* d_in, const int* in_sizes, int n_in,
                              void* d_out, int out_size, void* d_ws, size_t ws_size,
                              hipStream_t stream) {
  const float* x    = (const float*)d_in[0];
  const float* adj  = (const float*)d_in[1];
  const float* W[3]  = {(const float*)d_in[2], (const float*)d_in[6], (const float*)d_in[10]};
  const float* AS[3] = {(const float*)d_in[3], (const float*)d_in[7], (const float*)d_in[11]};
  const float* AD[3] = {(const float*)d_in[4], (const float*)d_in[8], (const float*)d_in[12]};
  const float* BI[3] = {(const float*)d_in[5], (const float*)d_in[9], (const float*)d_in[13]};
  const float* pw = (const float*)d_in[14];
  const float* pb = (const float*)d_in[15];
  float* out = (float*)d_out;

  // workspace layout
  char* ws = (char*)d_ws;
  size_t o = 0;
  __bf16* xbf   = (__bf16*)(ws + o); o += (size_t)BB * NN * FF * 2;        // 2 MB
  __bf16* wbfT  = (__bf16*)(ws + o); o += (size_t)HH * FF * FF * 2;        // 256 KB
  __bf16* hpbfT = (__bf16*)(ws + o); o += (size_t)BB * HH * NN * FF * 2;   // 16.8 MB
  float*  sb    = (float*)(ws + o);  o += (size_t)BB * HH * NN * 4;
  float*  db    = (float*)(ws + o);  o += (size_t)BB * HH * NN * 4;
  float*  rmax  = (float*)(ws + o);  o += (size_t)BB * HH * NN * 4;
  float*  rsum  = (float*)(ws + o);  o += (size_t)BB * HH * NN * 4;
  float*  obhT  = (float*)(ws + o);  o += (size_t)BB * HH * NN * FF * 4;   // 33.5 MB
  float*  x1    = (float*)(ws + o);  o += (size_t)BB * NN * FF * 4;
  float*  x2    = (float*)(ws + o);  o += (size_t)BB * NN * FF * 4;
  float*  x3    = (float*)(ws + o);  o += (size_t)BB * NN * FF * 4;
  float*  pooled = (float*)(ws + o); o += (size_t)BB * PRED_IN * 4;
  float*  XL[3] = {x1, x2, x3};

  const int nx = BB * NN * FF;      // 1,048,576
  const int nw = HH * FF * FF;      // 131,072

  cast_f32_bf16_kernel<<<(nx + 255) / 256, 256, 0, stream>>>(x, xbf, nx);

  for (int l = 0; l < 3; ++l) {
    cast_w_transpose_kernel<<<(nw + 255) / 256, 256, 0, stream>>>(W[l], wbfT);
    gemm_xw_kernel<<<8192, 128, 0, stream>>>(xbf, wbfT, hpbfT);
    sd_kernel<<<(BB * HH * NN + 255) / 256, 256, 0, stream>>>(hpbfT, AS[l], AD[l], sb, db);
    rowstats_kernel<<<8192, 256, 0, stream>>>(sb, db, adj, rmax, rsum);
    aggregate_kernel<<<1024, 128, 0, stream>>>(sb, db, rmax, rsum, adj, hpbfT, obhT);
    headmean_kernel<<<(nx + 255) / 256, 256, 0, stream>>>(obhT, BI[l], XL[l], xbf,
                                                          (l < 2) ? 1 : 0);
  }

  pool_kernel<<<(BB * PRED_IN + 255) / 256, 256, 0, stream>>>(x1, x2, x3, pooled);
  predict_kernel<<<1, 32, 0, stream>>>(pooled, pw, pb, out);
}
